// K_Means_46737834115148
// MI455X (gfx1250) — matrix operand order repaired and checked by reading, not compiled, not run on hardware
//
#include <hip/hip_runtime.h>
#include <hip/hip_bf16.h>

typedef __attribute__((ext_vector_type(16))) __bf16 v16bf;
typedef __attribute__((ext_vector_type(8)))  float  v8f;

#define N_ROWS 131072
#define DIM    256
#define KC     512
#define ROWS_PER_BLOCK 128
#define TILE_N 16
#define NTILES (KC / TILE_N)
#define LDS_STRIDE 264   // bf16 elements per staged row (256 + 8 pad -> conflict-free)

static __device__ __forceinline__ unsigned short f2bf_bits(float f) {
    unsigned u = __builtin_bit_cast(unsigned, f);
    unsigned r = u + 0x7fffu + ((u >> 16) & 1u);   // round-to-nearest-even
    return (unsigned short)(r >> 16);
}
static __device__ __forceinline__ __bf16 f2bf(float f) {
    unsigned short h = f2bf_bits(f);
    return __builtin_bit_cast(__bf16, h);
}

// ---------------------------------------------------------------------------
// Kernel 0: zero scatter accumulators (sums + counts) in ws, zero out-scalar.
// ---------------------------------------------------------------------------
__global__ void kinit(float* __restrict__ ws, int total, float* __restrict__ sumOut) {
    int i = blockIdx.x * blockDim.x + threadIdx.x;
    if (i < total) ws[i] = 0.0f;
    if (i == 0)    *sumOut = 0.0f;
}

// ---------------------------------------------------------------------------
// Kernel 0b: per-centroid prep: c_sq[k] (fp32, exact) + bf16 copy of centroids
// so the hot kernel never converts in-loop.
// ---------------------------------------------------------------------------
__global__ void kprep(const float* __restrict__ cent,
                      float* __restrict__ csq,
                      unsigned short* __restrict__ centBf) {
    __shared__ float red[256];
    int k = blockIdx.x;
    float v = cent[(size_t)k * DIM + threadIdx.x];
    centBf[(size_t)k * DIM + threadIdx.x] = f2bf_bits(v);
    red[threadIdx.x] = v * v;
    __syncthreads();
    for (int s = 128; s > 0; s >>= 1) {
        if (threadIdx.x < s) red[threadIdx.x] += red[threadIdx.x + s];
        __syncthreads();
    }
    if (threadIdx.x == 0) csq[k] = red[0];
}

// ---------------------------------------------------------------------------
// Kernel 1: fused distances (bf16 WMMA cross term) + argmin + sum of min dist.
// Block = 256 threads = 8 waves; each wave owns a 16-row x tile; block loops
// over 32 centroid tiles of 16, double-buffered in LDS (one barrier/tile).
// ---------------------------------------------------------------------------
__global__ void __launch_bounds__(256) kassign(
    const float* __restrict__ x,
    const unsigned short* __restrict__ centBf,
    const float* __restrict__ csq, float* __restrict__ assignOut,
    float* __restrict__ sumOut)
{
    __shared__ __bf16 ldsC[2][TILE_N * LDS_STRIDE];
    __shared__ float  csqLds[KC];
    __shared__ float  red[256];

    const int tid  = threadIdx.x;
    const int wave = tid >> 5;
    const int lane = tid & 31;
    const int l15  = lane & 15;
    const int half = lane >> 4;
    const int rowBase = blockIdx.x * ROWS_PER_BLOCK + wave * 16;

    // Stage all c_sq into LDS once (L2-hot, 2KB).
    csqLds[tid]       = csq[tid];
    csqLds[tid + 256] = csq[tid + 256];

    // ---- Load A fragments once (ISA 16-bit A 16x32 layout), 8 chunks of K=32.
    // Lane L supplies row m = L&15; lanes m and m+16 jointly cover all 256 dims,
    // so accumulate an x^2 partial from the fp32 source values as we convert.
    v16bf afrag[8];
    float xsqPart = 0.0f;
    const float* xrow = x + (size_t)(rowBase + l15) * DIM;
    for (int c = 0; c < 8; ++c) {
        union { v16bf v; __bf16 h[16]; } ua;
        #pragma unroll
        for (int v = 0; v < 8; ++v) {
            int k = c * 32 + ((v >> 2) << 4) + (half << 3) + ((v & 3) << 1);
            float f0 = xrow[k], f1 = xrow[k + 1];
            xsqPart += f0 * f0 + f1 * f1;
            ua.h[2 * v]     = f2bf(f0);
            ua.h[2 * v + 1] = f2bf(f1);
        }
        afrag[c] = ua.v;
    }

    float minD[8];
    int   argK[8];
    #pragma unroll
    for (int v = 0; v < 8; ++v) { minD[v] = 3.4e38f; argK[v] = 0; }

    // Staging addresses for this thread: 32 bytes (16 bf16) of one tile row.
    const int stg_n  = tid >> 4;
    const int stg_ks = (tid & 15) << 4;

    // Prologue: stage tile 0 into buffer 0; near-prefetch tile 1.
    {
        const uint4* src = (const uint4*)(centBf + ((size_t)stg_n * DIM + stg_ks));
        uint4 q0 = src[0], q1 = src[1];
        uint4* dst = (uint4*)&ldsC[0][stg_n * LDS_STRIDE + stg_ks];
        dst[0] = q0; dst[1] = q1;
        __builtin_prefetch(centBf + ((size_t)(TILE_N + stg_n) * DIM + stg_ks), 0, 3);
    }

    for (int t = 0; t < NTILES; ++t) {
        __syncthreads();   // stage(t) visible; compute(t-1) done -> buf reuse safe
        const int centBase = t * TILE_N;

        // Stage tile t+1 into the other buffer; near-prefetch tile t+2.
        if (t + 1 < NTILES) {
            const uint4* src = (const uint4*)(centBf +
                ((size_t)(centBase + TILE_N + stg_n) * DIM + stg_ks));
            uint4 q0 = src[0], q1 = src[1];
            uint4* dst = (uint4*)&ldsC[(t + 1) & 1][stg_n * LDS_STRIDE + stg_ks];
            dst[0] = q0; dst[1] = q1;
            if (t + 2 < NTILES)
                __builtin_prefetch(centBf +
                    ((size_t)(centBase + 2 * TILE_N + stg_n) * DIM + stg_ks), 0, 3);
        }

        // cross[m][n] over full D via 8 x wmma_f32_16x16x32_bf16, two
        // independent accumulator chains for ILP.
        const __bf16* buf = ldsC[t & 1];
        v8f accA = {0.f, 0.f, 0.f, 0.f, 0.f, 0.f, 0.f, 0.f};
        v8f accB = {0.f, 0.f, 0.f, 0.f, 0.f, 0.f, 0.f, 0.f};
        #pragma unroll
        for (int c = 0; c < 4; ++c) {
            // B layout: n = lane&15, lanes 0-15 hold K=0..7,16..23, lanes 16-31 K=8..15,24..31
            union { v16bf v; uint4 q[2]; } ubA, ubB;
            const uint4* pA = (const uint4*)&buf[l15 * LDS_STRIDE + c * 32 + (half << 3)];
            const uint4* pB = (const uint4*)&buf[l15 * LDS_STRIDE + (c + 4) * 32 + (half << 3)];
            ubA.q[0] = pA[0]; ubA.q[1] = pA[2];
            ubB.q[0] = pB[0]; ubB.q[1] = pB[2];
            accA = __builtin_amdgcn_wmma_f32_16x16x32_bf16(
                false, afrag[c],     false, ubA.v, (short)0, accA, false, false);
            accB = __builtin_amdgcn_wmma_f32_16x16x32_bf16(
                false, afrag[c + 4], false, ubB.v, (short)0, accB, false, false);
        }

        // dist(m,n) - x_sq = c_sq[n] - 2*cross ; C/D layout: n=lane&15, m=v+8*half.
        // Candidate index strictly increases with t per lane, so strict '<'
        // preserves first-minimum semantics (no tie-break needed here).
        float cs = csqLds[centBase + l15];
        int   a  = centBase + l15;
        #pragma unroll
        for (int v = 0; v < 8; ++v) {
            float d = cs - 2.0f * (accA[v] + accB[v]);
            if (d < minD[v]) { minD[v] = d; argK[v] = a; }
        }
    }

    // Full x^2 per row (combine lanes L and L+16).
    float xsq = xsqPart + __shfl_xor(xsqPart, 16, 32);

    // Butterfly argmin across the 16 lanes of each half (n-residues of a row);
    // here different lanes hold different n, so break ties on smaller index.
    #pragma unroll
    for (int off = 8; off >= 1; off >>= 1) {
        #pragma unroll
        for (int v = 0; v < 8; ++v) {
            float od = __shfl_xor(minD[v], off, 32);
            int   oa = __shfl_xor(argK[v], off, 32);
            if (od < minD[v] || (od == minD[v] && oa < argK[v])) { minD[v] = od; argK[v] = oa; }
        }
    }

    // Per-half sum over its 8 rows of (minDist + x_sq); fetch x_sq for row
    // m = v + 8*half from the lane that owns it (uniform shuffles).
    float rowSum = 0.0f;
    #pragma unroll
    for (int v = 0; v < 8; ++v) {
        float xs = __shfl(xsq, v + (half << 3), 16);
        rowSum += minD[v] + xs;
    }

    if (l15 == 0) {
        #pragma unroll
        for (int v = 0; v < 8; ++v)
            assignOut[rowBase + v + (half << 3)] = (float)argK[v];
    }

    __syncthreads();   // done with last tile before reusing shared scratch
    red[tid] = (l15 == 0) ? rowSum : 0.0f;
    __syncthreads();
    for (int s = 128; s > 0; s >>= 1) {
        if (tid < s) red[tid] += red[tid + s];
        __syncthreads();
    }
    if (tid == 0) atomicAdd(sumOut, red[0]);
}

// ---------------------------------------------------------------------------
// Kernel 2: segment-sum scatter: sums[a] += x[row], counts[a] += 1
// ---------------------------------------------------------------------------
__global__ void __launch_bounds__(256) kscatter(
    const float* __restrict__ x, const float* __restrict__ assign,
    float* __restrict__ sums, float* __restrict__ counts)
{
    int rowStart = blockIdx.x * ROWS_PER_BLOCK;
    for (int r = 0; r < ROWS_PER_BLOCK; ++r) {
        int row = rowStart + r;
        int a   = (int)assign[row];
        atomicAdd(&sums[(size_t)a * DIM + threadIdx.x],
                  x[(size_t)row * DIM + threadIdx.x]);
        if (threadIdx.x == 0) atomicAdd(&counts[a], 1.0f);
    }
}

// ---------------------------------------------------------------------------
// Kernel 3: updated_centroids = sums / counts  (0/0 -> NaN, matches reference)
// ---------------------------------------------------------------------------
__global__ void kfinal(const float* __restrict__ sums,
                       const float* __restrict__ counts,
                       float* __restrict__ out)
{
    int i = blockIdx.x * blockDim.x + threadIdx.x;
    if (i < KC * DIM) out[i] = sums[i] / counts[i >> 8];
}

// ---------------------------------------------------------------------------
extern "C" void kernel_launch(void* const* d_in, const int* in_sizes, int n_in,
                              void* d_out, int out_size, void* d_ws, size_t ws_size,
                              hipStream_t stream) {
    (void)in_sizes; (void)n_in; (void)out_size; (void)ws_size;
    const float* x    = (const float*)d_in[0];
    const float* cent = (const float*)d_in[1];

    float* out       = (float*)d_out;
    float* assignOut = out;                       // [N]      (indices as floats)
    float* centOut   = out + N_ROWS;              // [K*D]
    float* sumOut    = out + N_ROWS + KC * DIM;   // scalar

    float* ws     = (float*)d_ws;
    float* sums   = ws;                    // [K*D] f32
    float* counts = ws + KC * DIM;         // [K]   f32
    float* csq    = ws + KC * DIM + KC;    // [K]   f32
    unsigned short* centBf = (unsigned short*)(ws + KC * DIM + 2 * KC); // [K*D] bf16

    int zTotal = KC * DIM + KC;            // sums + counts
    kinit<<<(zTotal + 255) / 256, 256, 0, stream>>>(ws, zTotal, sumOut);
    kprep<<<KC, 256, 0, stream>>>(cent, csq, centBf);
    kassign<<<N_ROWS / ROWS_PER_BLOCK, 256, 0, stream>>>(x, centBf, csq, assignOut, sumOut);
    kscatter<<<N_ROWS / ROWS_PER_BLOCK, 256, 0, stream>>>(x, assignOut, sums, counts);
    kfinal<<<(KC * DIM + 255) / 256, 256, 0, stream>>>(sums, counts, centOut);
}
